// ADMDecoder_4063039062760
// MI455X (gfx1250) — compile-verified, matrix operand order checked by reading, not run on hardware
//
#include <hip/hip_runtime.h>
#include <hip/hip_bf16.h>

// ---------------------------------------------------------------------------
// MI455X (gfx1250, wave32) implementation of the ADM decoder forward pass.
// Dense projections + pair RBF/MLP on v_wmma_f32_16x16x32_f16; A-tiles moved
// global->LDS by the Tensor Data Mover (TDM) when available; fragments built
// from LDS with 16B ds_load_b128; uniform fast-path staging (no divergent
// guarded loads in the hot path).
// ---------------------------------------------------------------------------

typedef __attribute__((ext_vector_type(16))) _Float16 v16h;
typedef __attribute__((ext_vector_type(8)))  _Float16 v8h;
typedef __attribute__((ext_vector_type(4)))  _Float16 v4h;
typedef __attribute__((ext_vector_type(8)))  float    v8f;
typedef __attribute__((ext_vector_type(4)))  unsigned int v4u;
typedef __attribute__((ext_vector_type(8)))  int      v8i;
typedef __attribute__((ext_vector_type(4)))  int      v4i;

#if defined(__has_builtin)
#if __has_builtin(__builtin_amdgcn_tensor_load_to_lds) && \
    __has_builtin(__builtin_amdgcn_s_wait_tensorcnt)
#define HAVE_TDM 1
#endif
#endif
#ifndef HAVE_TDM
#define HAVE_TDM 0
#endif

#define NNODE 8192
#define KNB   16

__device__ __forceinline__ float geluf(float x) {
  const float c = 0.7978845608028654f;
  return 0.5f * x * (1.0f + tanhf(c * (x + 0.044715f * x * x * x)));
}
__device__ __forceinline__ float softplusf(float x) {
  return (x > 20.f) ? x : log1pf(expf(x));
}
// Build a WMMA 16-bit fragment from f16 LDS: elems 0..7 at p, 8..15 at p+16.
__device__ __forceinline__ v16h ldfrag(const _Float16* p) {
  v8h lo = *(const v8h*)p;
  v8h hi = *(const v8h*)(p + 16);
  return __builtin_shufflevector(lo, hi, 0, 1, 2, 3, 4, 5, 6, 7,
                                 8, 9, 10, 11, 12, 13, 14, 15);
}
// Build a WMMA 16-bit fragment from f32 LDS (converts at read time).
__device__ __forceinline__ v16h ldfrag32(const float* p) {
  v8f lo = *(const v8f*)p;
  v8f hi = *(const v8f*)(p + 16);
  v16h f;
#pragma unroll
  for (int i = 0; i < 8; i++) {
    f[i] = (_Float16)lo[i];
    f[i + 8] = (_Float16)hi[i];
  }
  return f;
}

// Frames: Rm[i*3+j] = e_j[i] (columns e1,e2,e3), tv = CA (unscaled).
__device__ void make_frame(const float* __restrict__ p, float* Rm, float* tv) {
  float nx = p[0], ny = p[1], nz = p[2];
  float cax = p[3], cay = p[4], caz = p[5];
  float cx = p[6], cy = p[7], cz = p[8];
  float e1x = cx - cax, e1y = cy - cay, e1z = cz - caz;
  float r = rsqrtf(e1x * e1x + e1y * e1y + e1z * e1z + 1e-8f);
  e1x *= r; e1y *= r; e1z *= r;
  float ux = nx - cax, uy = ny - cay, uz = nz - caz;
  float dp = ux * e1x + uy * e1y + uz * e1z;
  float e2x = ux - dp * e1x, e2y = uy - dp * e1y, e2z = uz - dp * e1z;
  r = rsqrtf(e2x * e2x + e2y * e2y + e2z * e2z + 1e-8f);
  e2x *= r; e2y *= r; e2z *= r;
  float e3x = e1y * e2z - e1z * e2y;
  float e3y = e1z * e2x - e1x * e2z;
  float e3z = e1x * e2y - e1y * e2x;
  Rm[0] = e1x; Rm[1] = e2x; Rm[2] = e3x;
  Rm[3] = e1y; Rm[4] = e2y; Rm[5] = e3y;
  Rm[6] = e1z; Rm[7] = e2z; Rm[8] = e3z;
  tv[0] = cax; tv[1] = cay; tv[2] = caz;
}

// ---------------------------------------------------------------------------
// Generic GEMM: C[M,Nd] = act(A[M,Kd] @ W[Kd,Nd] + bias) (+ resid)
// Requirements (all call sites satisfy them): M % 64 == 0, Kd % 16 == 0.
// 128 threads = 4 waves; block tile 64(M) x 64(N); WMMA f32_16x16x32_f16.
// A chunk staged as raw fp32 (TDM when available, float4 copies otherwise);
// B chunk staged as raw fp32 with a block-uniform fast path; out-of-range
// k tail in A is cancelled by zeroed weight columns in sB32.
// ---------------------------------------------------------------------------
__global__ __launch_bounds__(128) void k_gemm(
    const float* __restrict__ A, const float* __restrict__ W,
    const float* __restrict__ bias, const float* __restrict__ resid,
    float* __restrict__ C, int M, int Kd, int Nd, int act) {
  __shared__ float sA32[64 * 32];    // [row][k], fp32
  __shared__ float sB32[64 * 32];    // [colLocal][k], transposed, fp32
  const int tid = threadIdx.x, wid = tid >> 5, lane = tid & 31;
  const int hlf = (lane >> 4) & 1, mm = lane & 15, nn = lane & 15;
  const int mBase = blockIdx.x * 64, nBase = blockIdx.y * 64;
  const bool nFull = (nBase + 64 <= Nd);
  v8f acc[4] = {};
  for (int k0 = 0; k0 < Kd; k0 += 32) {
    // Stage W chunk transposed: sB32[cl][kk]; coalesced along columns.
    if (nFull && (k0 + 32 <= Kd)) {
      const float* Wt = W + (size_t)k0 * Nd + nBase;
#pragma unroll
      for (int t = 0; t < 16; t++) {
        int i = tid + t * 128;          // 0..2047
        int kk = i >> 6, cl = i & 63;
        sB32[cl * 32 + kk] = Wt[kk * Nd + cl];
      }
    } else {
#pragma unroll
      for (int t = 0; t < 16; t++) {
        int i = tid + t * 128;
        int kk = i >> 6, cl = i & 63;
        int gk = k0 + kk, gc = nBase + cl;
        int a0 = gk < Kd ? gk : 0;
        int a1 = gc < Nd ? gc : 0;
        float sel = (gk < Kd && gc < Nd) ? 1.f : 0.f;
        sB32[cl * 32 + kk] = W[(size_t)a0 * Nd + a1] * sel;
      }
    }
#if HAVE_TDM
    if (wid == 0) {
      // Tensor DMA: 64x32 fp32 tile of A (rows mBase.., k columns k0..k0+31).
      unsigned ldsOff = (unsigned)(unsigned long long)(void*)sA32;
      unsigned long long ga =
          (unsigned long long)(const void*)(A + (size_t)mBase * Kd + k0);
      v4u g0;
      g0[0] = 1u;                                   // count=1, user mode
      g0[1] = ldsOff;                               // lds_addr (bytes)
      g0[2] = (unsigned)(ga & 0xffffffffu);         // global_addr[31:0]
      g0[3] = (unsigned)((ga >> 32) & 0x1ffffffu) | (2u << 30);  // type=2
      v8i g1;
      g1[0] = (int)(2u << 16);                      // data_size = 4B
      g1[1] = (int)(((unsigned)Kd & 0xffffu) << 16);        // tdim0[15:0]
      g1[2] = (int)(((unsigned)Kd >> 16) |
                    (((unsigned)M & 0xffffu) << 16));       // tdim0 hi, tdim1 lo
      g1[3] = (int)(((unsigned)M >> 16) | (32u << 16));     // tdim1 hi, tile0=32
      g1[4] = 64;                                   // tile_dim1=64, tile_dim2=0
      g1[5] = Kd;                                   // dim0_stride[31:0]
      g1[6] = 0;                                    // stride hi bits
      g1[7] = 0;
      v4i g2 = {0, 0, 0, 0};
      v4i g3 = {0, 0, 0, 0};
      v8i g4 = {0, 0, 0, 0, 0, 0, 0, 0};
      __builtin_amdgcn_tensor_load_to_lds(g0, g1, g2, g3, g4, 0);
      __builtin_amdgcn_s_wait_tensorcnt(0);
    }
#else
    // Fallback: raw float4 copies (no conversion, no selects).
#pragma unroll
    for (int t = 0; t < 4; t++) {
      int j = tid + t * 128;          // 0..511 float4 groups
      int r = j >> 3, c4 = j & 7;
      int gc = k0 + c4 * 4;
      int cgc = gc < Kd ? gc : 0;
      float4 v = *(const float4*)(A + (size_t)(mBase + r) * Kd + cgc);
      *(float4*)(sA32 + r * 32 + c4 * 4) = v;
    }
#endif
    __syncthreads();
    if (k0 + 32 < Kd)
      __builtin_prefetch(&A[(size_t)mBase * Kd + k0 + 32], 0, 1);
    v16h af = ldfrag32(sA32 + (wid * 16 + mm) * 32 + hlf * 8);
#pragma unroll
    for (int t = 0; t < 4; t++) {
      v16h bf = ldfrag32(sB32 + (t * 16 + nn) * 32 + hlf * 8);
      acc[t] = __builtin_amdgcn_wmma_f32_16x16x32_f16(
          false, af, false, bf, (short)0, acc[t], false, false);
    }
    __syncthreads();
  }
#pragma unroll
  for (int t = 0; t < 4; t++) {
    int col = nBase + t * 16 + nn;
    if (col >= Nd) continue;
#pragma unroll
    for (int j = 0; j < 8; j++) {
      int row = mBase + wid * 16 + j + (hlf << 3);
      float v = acc[t][j];
      if (bias) v += bias[col];
      if (act == 1) v = geluf(v);
      if (resid) v += resid[(size_t)row * Nd + col];
      C[(size_t)row * Nd + col] = v;
    }
  }
}

// ---------------------------------------------------------------------------
// Pair geometry: relpos one-hot, dirs (15), rot (9) -> PAIR[p][64], PM[p].
// ---------------------------------------------------------------------------
__global__ __launch_bounds__(64) void k_pair_geom(
    const float* __restrict__ pos, const int* __restrict__ nbr,
    const int* __restrict__ resi, const int* __restrict__ chain,
    const unsigned char* __restrict__ mask,
    const float* __restrict__ w_rel, const float* __restrict__ w_dir,
    const float* __restrict__ w_rot, float* __restrict__ PAIR,
    float* __restrict__ PM) {
  int p = blockIdx.x, t = threadIdx.x;
  int n = p >> 4;
  __shared__ float sDir[15], sRot[9], sSame;
  __shared__ int sRel;
  if (t == 0) {
    int m = nbr[p];
    float Rn[9], tn[3], Rb[9], tb[3];
    make_frame(pos + (size_t)n * 15, Rn, tn);
    make_frame(pos + (size_t)m * 15, Rb, tb);
    for (int a = 0; a < 5; a++) {
      float vx = pos[(size_t)m * 15 + a * 3 + 0] - tn[0];
      float vy = pos[(size_t)m * 15 + a * 3 + 1] - tn[1];
      float vz = pos[(size_t)m * 15 + a * 3 + 2] - tn[2];
      float d0 = Rn[0] * vx + Rn[3] * vy + Rn[6] * vz;
      float d1 = Rn[1] * vx + Rn[4] * vy + Rn[7] * vz;
      float d2 = Rn[2] * vx + Rn[5] * vy + Rn[8] * vz;
      float r = rsqrtf(d0 * d0 + d1 * d1 + d2 * d2 + 1e-8f);
      sDir[a * 3 + 0] = d0 * r; sDir[a * 3 + 1] = d1 * r; sDir[a * 3 + 2] = d2 * r;
    }
    for (int i = 0; i < 3; i++)
      for (int l = 0; l < 3; l++)
        sRot[i * 3 + l] = Rn[0 + i] * Rb[0 + l] + Rn[3 + i] * Rb[3 + l] +
                          Rn[6 + i] * Rb[6 + l];
    int rl = resi[m] - resi[n];
    rl = (rl < -32) ? -32 : ((rl > 32) ? 32 : rl);
    sRel = rl + 32;
    sSame = (chain[n] == chain[m]) ? 1.f : 0.f;
    PM[p] = (mask[n] && mask[m] && (m >= 0)) ? 1.f : 0.f;
  }
  __syncthreads();
  float acc = sSame * (w_rel[sRel * 64 + t] + w_rel[65 * 64 + t]);
  for (int i = 0; i < 15; i++) acc += sDir[i] * w_dir[i * 64 + t];
  for (int i = 0; i < 9; i++) acc += sRot[i] * w_rot[i * 64 + t];
  PAIR[(size_t)p * 64 + t] = acc;
}

// ---------------------------------------------------------------------------
// Pair RBF: generate 400 RBF features in LDS (fp16), WMMA against w_dist,
// accumulate into PAIR.  64 pairs per block (4 waves x 16 rows).
// ---------------------------------------------------------------------------
__global__ __launch_bounds__(128) void k_pair_rbf(
    const float* __restrict__ pos, const int* __restrict__ nbr,
    const float* __restrict__ w_dist, float* __restrict__ PAIR) {
  __shared__ _Float16 sF[64 * 416];
  __shared__ float sScr[1600];           // distances; reused as 64x32 f16 W tile
  int tid = threadIdx.x;
  int pBase = blockIdx.x * 64;
  for (int i = tid; i < 64 * 25; i += 128) {
    int pl = i / 25, q = i % 25;
    int p = pBase + pl;
    int n = p >> 4;
    int m = nbr[p];
    int anb = q / 5, asf = q % 5;
    float dx = pos[(size_t)m * 15 + anb * 3 + 0] - pos[(size_t)n * 15 + asf * 3 + 0];
    float dy = pos[(size_t)m * 15 + anb * 3 + 1] - pos[(size_t)n * 15 + asf * 3 + 1];
    float dz = pos[(size_t)m * 15 + anb * 3 + 2] - pos[(size_t)n * 15 + asf * 3 + 2];
    sScr[i] = sqrtf(dx * dx + dy * dy + dz * dz + 1e-8f);
  }
  __syncthreads();
  for (int i = tid; i < 64 * 416; i += 128) {
    int pl = i / 416, f = i % 416;
    float v = 0.f;
    if (f < 400) {
      int a2 = f >> 4, b = f & 15;
      float d = sScr[pl * 25 + a2];
      float c = 22.f * b / 15.f;           // linspace(0,22,16)
      float z = (d - c) * (1.f / 1.375f);  // sigma = 22/16
      v = expf(-z * z);
    }
    sF[pl * 416 + f] = (_Float16)v;
  }
  int wid = tid >> 5, lane = tid & 31, hlf = (lane >> 4) & 1;
  int mm = lane & 15, nn = lane & 15;
  _Float16* sB = (_Float16*)sScr;          // 64 cols x 32 k, 4KB
  v8f acc[4] = {};
  for (int k0 = 0; k0 < 416; k0 += 32) {
    __syncthreads();
    if (k0 + 32 <= 400) {
      const float* wt = w_dist + (size_t)k0 * 64;
#pragma unroll
      for (int t = 0; t < 16; t++) {
        int i = tid + t * 128;
        int kk = i >> 6, cl = i & 63;
        sB[cl * 32 + kk] = (_Float16)wt[kk * 64 + cl];
      }
    } else {
#pragma unroll
      for (int t = 0; t < 16; t++) {
        int i = tid + t * 128;
        int kk = i >> 6, cl = i & 63;
        int gk = k0 + kk;
        int a0 = gk < 400 ? gk : 0;
        float sel = gk < 400 ? 1.f : 0.f;
        sB[cl * 32 + kk] = (_Float16)(w_dist[(size_t)a0 * 64 + cl] * sel);
      }
    }
    __syncthreads();
    v16h af = ldfrag(sF + (wid * 16 + mm) * 416 + k0 + hlf * 8);
#pragma unroll
    for (int t = 0; t < 4; t++) {
      v16h bf = ldfrag(sB + (t * 16 + nn) * 32 + hlf * 8);
      acc[t] = __builtin_amdgcn_wmma_f32_16x16x32_f16(
          false, af, false, bf, (short)0, acc[t], false, false);
    }
  }
#pragma unroll
  for (int t = 0; t < 4; t++) {
    int col = t * 16 + nn;
#pragma unroll
    for (int j = 0; j < 8; j++) {
      int row = wid * 16 + j + (hlf << 3);
      PAIR[(size_t)(pBase + row) * 64 + col] += acc[t][j];
    }
  }
}

// ---------------------------------------------------------------------------
// Pair LN + MLP 64->128(gelu)->64 fused in LDS with WMMA. 64 pairs / block.
// ---------------------------------------------------------------------------
__global__ __launch_bounds__(128) void k_pair_mlp(
    float* __restrict__ PAIR, const float* __restrict__ ln_s,
    const float* __restrict__ ln_b, const float* __restrict__ w1,
    const float* __restrict__ b1, const float* __restrict__ w2,
    const float* __restrict__ b2) {
  __shared__ _Float16 sX[64 * 64];
  __shared__ _Float16 sH[64 * 128];
  __shared__ float sW32[128 * 32];        // [colLocal][k], fp32
  int tid = threadIdx.x;
  int pBase = blockIdx.x * 64;
  if (tid < 64) {
    size_t base = (size_t)(pBase + tid) * 64;
    float mean = 0.f;
    for (int d = 0; d < 64; d++) mean += PAIR[base + d];
    mean *= (1.f / 64.f);
    float var = 0.f;
    for (int d = 0; d < 64; d++) { float x = PAIR[base + d] - mean; var += x * x; }
    var *= (1.f / 64.f);
    float rs = rsqrtf(var + 1e-5f);
    for (int d = 0; d < 64; d++)
      sX[tid * 64 + d] = (_Float16)((PAIR[base + d] - mean) * rs * ln_s[d] + ln_b[d]);
  }
  int wid = tid >> 5, lane = tid & 31, hlf = (lane >> 4) & 1;
  int mm = lane & 15, nn = lane & 15;
  {
    v8f acc[8] = {};
    for (int k0 = 0; k0 < 64; k0 += 32) {
      __syncthreads();
      const float* w1t = w1 + (size_t)k0 * 128;
#pragma unroll
      for (int t = 0; t < 32; t++) {       // 128 cols x 32 k
        int i = tid + t * 128;
        int kk = i >> 7, cl = i & 127;
        sW32[cl * 32 + kk] = w1t[kk * 128 + cl];
      }
      __syncthreads();
      v16h af = ldfrag(sX + (wid * 16 + mm) * 64 + k0 + hlf * 8);
#pragma unroll
      for (int t = 0; t < 8; t++) {
        v16h bf = ldfrag32(sW32 + (t * 16 + nn) * 32 + hlf * 8);
        acc[t] = __builtin_amdgcn_wmma_f32_16x16x32_f16(
            false, af, false, bf, (short)0, acc[t], false, false);
      }
    }
#pragma unroll
    for (int t = 0; t < 8; t++) {
      int col = t * 16 + nn;
#pragma unroll
      for (int j = 0; j < 8; j++) {
        int row = wid * 16 + j + (hlf << 3);
        sH[row * 128 + col] = (_Float16)geluf(acc[t][j] + b1[col]);
      }
    }
  }
  {
    v8f acc[4] = {};
    for (int k0 = 0; k0 < 128; k0 += 32) {
      __syncthreads();
      const float* w2t = w2 + (size_t)k0 * 64;
#pragma unroll
      for (int t = 0; t < 16; t++) {       // 64 cols x 32 k
        int i = tid + t * 128;
        int kk = i >> 6, cl = i & 63;
        sW32[cl * 32 + kk] = w2t[kk * 64 + cl];
      }
      __syncthreads();
      v16h af = ldfrag(sH + (wid * 16 + mm) * 128 + k0 + hlf * 8);
#pragma unroll
      for (int t = 0; t < 4; t++) {
        v16h bf = ldfrag32(sW32 + (t * 16 + nn) * 32 + hlf * 8);
        acc[t] = __builtin_amdgcn_wmma_f32_16x16x32_f16(
            false, af, false, bf, (short)0, acc[t], false, false);
      }
    }
#pragma unroll
    for (int t = 0; t < 4; t++) {
      int col = t * 16 + nn;
#pragma unroll
      for (int j = 0; j < 8; j++) {
        int row = wid * 16 + j + (hlf << 3);
        PAIR[(size_t)(pBase + row) * 64 + col] = acc[t][j] + b2[col];
      }
    }
  }
}

// ---------------------------------------------------------------------------
// Small / elementwise kernels
// ---------------------------------------------------------------------------
__global__ void k_softmax_rows(const float* __restrict__ in, float* __restrict__ out,
                               int n, int d) {
  int r = blockIdx.x * blockDim.x + threadIdx.x;
  if (r >= n) return;
  float mx = -1e30f;
  for (int j = 0; j < d; j++) mx = fmaxf(mx, in[(size_t)r * d + j]);
  float s = 0.f;
  for (int j = 0; j < d; j++) s += expf(in[(size_t)r * d + j] - mx);
  float inv = 1.f / s;
  for (int j = 0; j < d; j++) out[(size_t)r * d + j] = expf(in[(size_t)r * d + j] - mx) * inv;
}

__global__ void k_embed(const float* __restrict__ xin, const int* __restrict__ aa,
                        const float* __restrict__ SMT, const float* __restrict__ w_aa,
                        const float* __restrict__ w_trunk, float* __restrict__ X,
                        float* __restrict__ INC) {
  int n = blockIdx.x, d = threadIdx.x;
  float v = xin[(size_t)n * 256 + d] + w_aa[aa[n] * 256 + d];
  for (int j = 0; j < 20; j++) v += SMT[n * 20 + j] * w_trunk[j * 256 + d];
  X[(size_t)n * 256 + d] = v;
  INC[(size_t)n * 256 + d] = v;
}

__global__ __launch_bounds__(32) void k_points(
    const float* __restrict__ pos, const float* __restrict__ QPr,
    const float* __restrict__ KPr, float* __restrict__ QG, float* __restrict__ KG) {
  int n = blockIdx.x, t = threadIdx.x;
  __shared__ float Rm[9], tv[3];
  if (t == 0) {
    make_frame(pos + (size_t)n * 15, Rm, tv);
    tv[0] *= 0.1f; tv[1] *= 0.1f; tv[2] *= 0.1f;  // pos/SIGMA, SIGMA=10
  }
  __syncthreads();
  size_t base = (size_t)n * 96 + t * 3;
  float qx = QPr[base], qy = QPr[base + 1], qz = QPr[base + 2];
  float kx = KPr[base], ky = KPr[base + 1], kz = KPr[base + 2];
  for (int i = 0; i < 3; i++) {
    QG[base + i] = Rm[i * 3 + 0] * qx + Rm[i * 3 + 1] * qy + Rm[i * 3 + 2] * qz + tv[i];
    KG[base + i] = Rm[i * 3 + 0] * kx + Rm[i * 3 + 1] * ky + Rm[i * 3 + 2] * kz + tv[i];
  }
}

__global__ __launch_bounds__(128) void k_attn(
    const float* __restrict__ Q, const float* __restrict__ KB,
    const float* __restrict__ V, const float* __restrict__ PAIR,
    const float* __restrict__ PM, const float* __restrict__ QG,
    const float* __restrict__ KG, const int* __restrict__ nbr,
    const float* __restrict__ wb, const float* __restrict__ gamma,
    float* __restrict__ CAT) {
  int n = blockIdx.x, tid = threadIdx.x;
  __shared__ float sQ[256], sQg[96], sP[16 * 64], sL[128], sAw[128], sPm[16];
  __shared__ int sNb[16];
  if (tid < 16) { sNb[tid] = nbr[n * 16 + tid]; sPm[tid] = PM[n * 16 + tid]; }
  for (int i = tid; i < 256; i += 128) sQ[i] = Q[(size_t)n * 256 + i];
  if (tid < 96) sQg[tid] = QG[(size_t)n * 96 + tid];
  for (int i = tid; i < 1024; i += 128) sP[i] = PAIR[(size_t)n * 16 * 64 + i];
  __syncthreads();
  {
    int k = tid >> 3, h = tid & 7;
    int m = sNb[k];
    const float4* kp = (const float4*)(KB + (size_t)m * 256 + h * 32);
    const float4* qp = (const float4*)(sQ + h * 32);
    float qk = 0.f;
#pragma unroll
    for (int d4 = 0; d4 < 8; d4++) {
      float4 kv = kp[d4];
      float4 qv = qp[d4];
      qk += qv.x * kv.x + qv.y * kv.y + qv.z * kv.z + qv.w * kv.w;
    }
    float pl = 0.f;
    for (int d = 0; d < 64; d++) pl += sP[k * 64 + d] * wb[d * 8 + h];
    float d2 = 0.f;
    for (int i = 0; i < 12; i++) {
      float df = sQg[h * 12 + i] - KG[(size_t)m * 96 + h * 12 + i];
      d2 += df * df;
    }
    float lg = qk * 0.17677669529663687f + pl - softplusf(gamma[h]) * d2 * 0.125f;
    if (!(sPm[k] > 0.f)) lg = -30000.f;
    sL[k * 8 + h] = lg;
  }
  __syncthreads();
  if (tid < 8) {
    float mx = -1e30f;
    for (int k = 0; k < 16; k++) mx = fmaxf(mx, sL[k * 8 + tid]);
    float s = 0.f;
    for (int k = 0; k < 16; k++) { float e = expf(sL[k * 8 + tid] - mx); sAw[k * 8 + tid] = e; s += e; }
    float inv = 1.f / s;
    for (int k = 0; k < 16; k++) sAw[k * 8 + tid] *= inv;
  }
  __syncthreads();
  for (int i = tid; i < 256; i += 128) {
    int h = i >> 5, d = i & 31;
    float acc = 0.f;
    for (int k = 0; k < 16; k++) acc += sAw[k * 8 + h] * V[(size_t)sNb[k] * 256 + h * 32 + d];
    CAT[(size_t)n * 768 + i] = acc;
  }
  for (int i = tid; i < 512; i += 128) {
    int h = i >> 6, pp = i & 63;
    float acc = 0.f;
    for (int k = 0; k < 16; k++) acc += sAw[k * 8 + h] * sP[k * 64 + pp];
    CAT[(size_t)n * 768 + 256 + i] = acc;
  }
}

__global__ void k_residual_ln(float* __restrict__ X, const float* __restrict__ U,
                              float* __restrict__ INC, const float* __restrict__ s,
                              const float* __restrict__ b) {
  int n = blockIdx.x * blockDim.x + threadIdx.x;
  if (n >= NNODE) return;
  size_t base = (size_t)n * 256;
  float mean = 0.f;
  for (int d = 0; d < 256; d++) mean += X[base + d] + U[base + d];
  mean *= (1.f / 256.f);
  float var = 0.f;
  for (int d = 0; d < 256; d++) { float t = X[base + d] + U[base + d] - mean; var += t * t; }
  var *= (1.f / 256.f);
  float rs = rsqrtf(var + 1e-5f);
  for (int d = 0; d < 256; d++) {
    float u = U[base + d];
    float t = X[base + d] + u;
    INC[base + d] += u;
    X[base + d] = (t - mean) * rs * s[d] + b[d];
  }
}

__global__ void k_pooled(const float* __restrict__ pos, const int* __restrict__ nbr,
                         const unsigned char* __restrict__ mask,
                         const float* __restrict__ X, float* __restrict__ XP) {
  int n = blockIdx.x * blockDim.x + threadIdx.x;
  if (n >= NNODE) return;
  float pooled[16];
#pragma unroll
  for (int b = 0; b < 16; b++) pooled[b] = 0.f;
  float csum = 0.f;
  float cax = pos[(size_t)n * 15 + 3], cay = pos[(size_t)n * 15 + 4], caz = pos[(size_t)n * 15 + 5];
  float mn = mask[n] ? 1.f : 0.f;
  for (int k = 0; k < 16; k++) {
    int m = nbr[n * 16 + k];
    float mf = mn * (mask[m] ? 1.f : 0.f);
    float dx = pos[(size_t)m * 15 + 3] - cax;
    float dy = pos[(size_t)m * 15 + 4] - cay;
    float dz = pos[(size_t)m * 15 + 5] - caz;
    float d = sqrtf(dx * dx + dy * dy + dz * dz + 1e-8f);
    csum += mf;
    for (int b = 0; b < 16; b++) {
      float c = 22.f * b / 15.f;
      float z = (d - c) * (1.f / 1.375f);
      pooled[b] += mf * expf(-z * z);
    }
  }
  float inv = 1.f / fmaxf(csum, 1.f);
  size_t ob = (size_t)n * 272;
  for (int d = 0; d < 256; d++) XP[ob + d] = X[(size_t)n * 256 + d];
  for (int b = 0; b < 16; b++) XP[ob + 256 + b] = pooled[b] * inv;
}

__global__ void k_maskz(float* __restrict__ T, const unsigned char* __restrict__ mask) {
  int i = blockIdx.x * blockDim.x + threadIdx.x;
  if (i >= NNODE * 256) return;
  if (!mask[i >> 8]) T[i] = 0.f;
}

__global__ void k_segsum(const float* __restrict__ LU, const int* __restrict__ batch,
                         const int* __restrict__ chain,
                         const unsigned char* __restrict__ mask, float* BS, float* CS,
                         float* BC, float* CC) {
  int n = blockIdx.x, d = threadIdx.x;
  float m = mask[n] ? 1.f : 0.f;
  float v = LU[(size_t)n * 512 + d] * m;
  atomicAdd(&BS[batch[n] * 512 + d], v);
  atomicAdd(&CS[chain[n] * 512 + d], v);
  if (d == 0) { atomicAdd(&BC[batch[n]], m); atomicAdd(&CC[chain[n]], m); }
}

__global__ void k_gate(const float* __restrict__ G, const float* __restrict__ LU,
                       const int* __restrict__ seg, const float* __restrict__ ssum,
                       const float* __restrict__ scnt, float* __restrict__ HID,
                       int mode) {
  int i = blockIdx.x * blockDim.x + threadIdx.x;
  if (i >= NNODE * 512) return;
  int n = i >> 9, d = i & 511;
  float g = G[i];
  float mval;
  if (mode == 2) mval = LU[i];
  else { int s = seg[n]; mval = ssum[s * 512 + d] / fmaxf(scnt[s], 1.f); }
  if (mode == 0) HID[i] = g * mval; else HID[i] += g * mval;
}

__global__ void k_final_ln(float* __restrict__ X, const float* __restrict__ INC,
                           const float* si, const float* bi, const float* so,
                           const float* bo) {
  int n = blockIdx.x * blockDim.x + threadIdx.x;
  if (n >= NNODE) return;
  size_t base = (size_t)n * 256;
  float mi = 0.f;
  for (int d = 0; d < 256; d++) mi += INC[base + d];
  mi *= (1.f / 256.f);
  float vi = 0.f;
  for (int d = 0; d < 256; d++) { float t = INC[base + d] - mi; vi += t * t; }
  vi *= (1.f / 256.f);
  float rsi = rsqrtf(vi + 1e-5f);
  float m2 = 0.f;
  for (int d = 0; d < 256; d++) {
    float li = (INC[base + d] - mi) * rsi * si[d] + bi[d];
    m2 += X[base + d] + li;
  }
  m2 *= (1.f / 256.f);
  float v2 = 0.f;
  for (int d = 0; d < 256; d++) {
    float li = (INC[base + d] - mi) * rsi * si[d] + bi[d];
    float t = X[base + d] + li - m2;
    v2 += t * t;
  }
  v2 *= (1.f / 256.f);
  float rs2 = rsqrtf(v2 + 1e-5f);
  for (int d = 0; d < 256; d++) {
    float li = (INC[base + d] - mi) * rsi * si[d] + bi[d];
    X[base + d] = (X[base + d] + li - m2) * rs2 * so[d] + bo[d];
  }
}

__global__ void k_logsoftmax(float* __restrict__ out, int n, int d) {
  int r = blockIdx.x * blockDim.x + threadIdx.x;
  if (r >= n) return;
  float mx = -1e30f;
  for (int j = 0; j < d; j++) mx = fmaxf(mx, out[(size_t)r * d + j]);
  float s = 0.f;
  for (int j = 0; j < d; j++) s += expf(out[(size_t)r * d + j] - mx);
  float lse = mx + logf(s);
  for (int j = 0; j < d; j++) out[(size_t)r * d + j] -= lse;
}

// ---------------------------------------------------------------------------
static inline void gemm(const float* A, const float* W, const float* bias,
                        const float* resid, float* C, int M, int Kd, int Nd,
                        int act, hipStream_t s) {
  dim3 g((unsigned)((M + 63) / 64), (unsigned)((Nd + 63) / 64));
  k_gemm<<<g, 128, 0, s>>>(A, W, bias, resid, C, M, Kd, Nd, act);
}

extern "C" void kernel_launch(void* const* d_in, const int* in_sizes, int n_in,
                              void* d_out, int out_size, void* d_ws, size_t ws_size,
                              hipStream_t stream) {
  (void)in_sizes; (void)n_in; (void)out_size; (void)ws_size;
  int ix = 0;
  const int* aa = (const int*)d_in[ix++];
  const float* aatr = (const float*)d_in[ix++];
  const float* xin = (const float*)d_in[ix++];
  const float* pos = (const float*)d_in[ix++];
  const int* nbr = (const int*)d_in[ix++];
  const int* resi = (const int*)d_in[ix++];
  const int* chain = (const int*)d_in[ix++];
  const int* batch = (const int*)d_in[ix++];
  const unsigned char* mask = (const unsigned char*)d_in[ix++];
  const float* w_aa = (const float*)d_in[ix++];
  const float* w_trunk = (const float*)d_in[ix++];
  struct BlockP {
    const float *w_rel, *w_dist, *w_dir, *w_rot, *ln_s, *ln_b, *p_w1, *p_b1, *p_w2, *p_b2;
    const float *wq, *wk, *wv, *wb, *wqp, *wkp, *gamma, *wo;
    const float *rd1s, *rd1b, *rd2s, *rd2b;
    const float *u_w1, *u_w2, *wu, *wgl, *wgc, *wgb, *u_wo;
  } B[2];
  for (int b = 0; b < 2; b++) {
    BlockP& q = B[b];
    q.w_rel = (const float*)d_in[ix++]; q.w_dist = (const float*)d_in[ix++];
    q.w_dir = (const float*)d_in[ix++]; q.w_rot = (const float*)d_in[ix++];
    q.ln_s = (const float*)d_in[ix++]; q.ln_b = (const float*)d_in[ix++];
    q.p_w1 = (const float*)d_in[ix++]; q.p_b1 = (const float*)d_in[ix++];
    q.p_w2 = (const float*)d_in[ix++]; q.p_b2 = (const float*)d_in[ix++];
    q.wq = (const float*)d_in[ix++]; q.wk = (const float*)d_in[ix++];
    q.wv = (const float*)d_in[ix++]; q.wb = (const float*)d_in[ix++];
    q.wqp = (const float*)d_in[ix++]; q.wkp = (const float*)d_in[ix++];
    q.gamma = (const float*)d_in[ix++]; q.wo = (const float*)d_in[ix++];
    q.rd1s = (const float*)d_in[ix++]; q.rd1b = (const float*)d_in[ix++];
    q.rd2s = (const float*)d_in[ix++]; q.rd2b = (const float*)d_in[ix++];
    q.u_w1 = (const float*)d_in[ix++]; q.u_w2 = (const float*)d_in[ix++];
    q.wu = (const float*)d_in[ix++]; q.wgl = (const float*)d_in[ix++];
    q.wgc = (const float*)d_in[ix++]; q.wgb = (const float*)d_in[ix++];
    q.u_wo = (const float*)d_in[ix++];
  }
  const float* lni_s = (const float*)d_in[ix++];
  const float* lni_b = (const float*)d_in[ix++];
  const float* lno_s = (const float*)d_in[ix++];
  const float* lno_b = (const float*)d_in[ix++];
  const float* head_w1 = (const float*)d_in[ix++];
  const float* head_w2 = (const float*)d_in[ix++];

  // Workspace layout (fp32)
  float* ws = (float*)d_ws;
  size_t off = 0;
  auto take = [&](size_t nf) { float* p = ws + off; off += nf; return p; };
  const size_t Nn = NNODE;
  float* X    = take(Nn * 256);
  float* INC  = take(Nn * 256);
  float* SMT  = take(Nn * 20);
  float* PAIR = take(Nn * 16 * 64);
  float* PM   = take(Nn * 16);
  float* Qb   = take(Nn * 256);
  float* Kbf  = take(Nn * 256);
  float* Vb   = take(Nn * 256);
  float* QG   = take(Nn * 96);
  float* KG   = take(Nn * 96);
  float* CAT  = take(Nn * 768);  // point staging / attn concat / 512-wide temp
  float* UPD  = take(Nn * 256);
  float* XP   = take(Nn * 272);
  float* TP   = take(Nn * 256);
  float* LU   = take(Nn * 512);
  float* HID  = take(Nn * 512);
  float* BS   = take(16 * 512);
  float* CS   = take(64 * 512);
  float* BC   = take(16);
  float* CC   = take(64);
  float* QPr = CAT;
  float* KPr = CAT + Nn * 96;
  float* G   = CAT;  // 512-wide scratch (t1 / gated / head h1)

  k_softmax_rows<<<(NNODE + 63) / 64, 64, 0, stream>>>(aatr, SMT, NNODE, 20);
  k_embed<<<NNODE, 256, 0, stream>>>(xin, aa, SMT, w_aa, w_trunk, X, INC);

  for (int b = 0; b < 2; b++) {
    const BlockP& P = B[b];
    // pair features
    k_pair_geom<<<NNODE * KNB, 64, 0, stream>>>(pos, nbr, resi, chain, mask,
                                                P.w_rel, P.w_dir, P.w_rot, PAIR, PM);
    k_pair_rbf<<<(NNODE * KNB) / 64, 128, 0, stream>>>(pos, nbr, P.w_dist, PAIR);
    k_pair_mlp<<<(NNODE * KNB) / 64, 128, 0, stream>>>(PAIR, P.ln_s, P.ln_b,
                                                       P.p_w1, P.p_b1, P.p_w2, P.p_b2);
    // attention
    gemm(X, P.wq, nullptr, nullptr, Qb, NNODE, 256, 256, 0, stream);
    gemm(X, P.wk, nullptr, nullptr, Kbf, NNODE, 256, 256, 0, stream);
    gemm(X, P.wv, nullptr, nullptr, Vb, NNODE, 256, 256, 0, stream);
    gemm(X, P.wqp, nullptr, nullptr, QPr, NNODE, 256, 96, 0, stream);
    gemm(X, P.wkp, nullptr, nullptr, KPr, NNODE, 256, 96, 0, stream);
    k_points<<<NNODE, 32, 0, stream>>>(pos, QPr, KPr, QG, KG);
    k_attn<<<NNODE, 128, 0, stream>>>(Qb, Kbf, Vb, PAIR, PM, QG, KG, nbr,
                                      P.wb, P.gamma, CAT);
    gemm(CAT, P.wo, nullptr, nullptr, UPD, NNODE, 768, 256, 0, stream);
    k_residual_ln<<<(NNODE + 63) / 64, 64, 0, stream>>>(X, UPD, INC, P.rd1s, P.rd1b);
    // update
    k_pooled<<<(NNODE + 63) / 64, 64, 0, stream>>>(pos, nbr, mask, X, XP);
    gemm(XP, P.u_w1, nullptr, nullptr, G, NNODE, 272, 512, 1, stream);
    gemm(G, P.u_w2, nullptr, nullptr, TP, NNODE, 512, 256, 0, stream);
    k_maskz<<<(NNODE * 256 + 255) / 256, 256, 0, stream>>>(TP, mask);
    gemm(X, P.wu, nullptr, nullptr, LU, NNODE, 256, 512, 0, stream);
    (void)hipMemsetAsync(BS, 0, (16 * 512 + 64 * 512 + 16 + 64) * sizeof(float), stream);
    k_segsum<<<NNODE, 512, 0, stream>>>(LU, batch, chain, mask, BS, CS, BC, CC);
    gemm(X, P.wgb, nullptr, nullptr, G, NNODE, 256, 512, 1, stream);
    k_gate<<<(NNODE * 512 + 255) / 256, 256, 0, stream>>>(G, LU, batch, BS, BC, HID, 0);
    gemm(X, P.wgc, nullptr, nullptr, G, NNODE, 256, 512, 1, stream);
    k_gate<<<(NNODE * 512 + 255) / 256, 256, 0, stream>>>(G, LU, chain, CS, CC, HID, 1);
    gemm(X, P.wgl, nullptr, nullptr, G, NNODE, 256, 512, 1, stream);
    k_gate<<<(NNODE * 512 + 255) / 256, 256, 0, stream>>>(G, LU, batch, BS, BC, HID, 2);
    gemm(HID, P.u_wo, nullptr, TP, UPD, NNODE, 512, 256, 0, stream);
    k_residual_ln<<<(NNODE + 63) / 64, 64, 0, stream>>>(X, UPD, INC, P.rd2s, P.rd2b);
  }

  k_final_ln<<<(NNODE + 63) / 64, 64, 0, stream>>>(X, INC, lni_s, lni_b, lno_s, lno_b);
  gemm(X, head_w1, nullptr, nullptr, G, NNODE, 256, 512, 1, stream);
  gemm(G, head_w2, nullptr, nullptr, (float*)d_out, NNODE, 512, 20, 0, stream);
  k_logsoftmax<<<(NNODE + 63) / 64, 64, 0, stream>>>((float*)d_out, NNODE, 20);
}
